// SimCLR_Loss_43516608643557
// MI455X (gfx1250) — matrix operand order repaired and checked by reading, not compiled, not run on hardware
//
#include <hip/hip_runtime.h>
#include <hip/hip_bf16.h>
#include <math.h>

typedef __attribute__((ext_vector_type(16))) _Float16 v16h;
typedef __attribute__((ext_vector_type(8)))  _Float16 v8h;
typedef __attribute__((ext_vector_type(8)))  float    v8f;

namespace {
constexpr int   kB             = 4096;
constexpr int   kN             = 8192;              // 2*B
constexpr int   kD             = 256;
constexpr int   kTiles         = kN / 16;           // 512 column tiles
constexpr int   kChunks        = 8;                 // column chunks per row tile
constexpr int   kTilesPerChunk = kTiles / kChunks;  // 64
constexpr float kInvT          = 2.0f;              // 1 / 0.5
constexpr float kEps           = 1e-8f;
}

// ---------------------------------------------------------------------------
// Kernel 1: row L2-normalize z=[z_i;z_j] -> f16 Zn in workspace.
// ---------------------------------------------------------------------------
__global__ __launch_bounds__(256) void simclr_normalize_f16(
    const float* __restrict__ zi, const float* __restrict__ zj,
    _Float16* __restrict__ zh) {
  const int row  = (blockIdx.x * blockDim.x + threadIdx.x) >> 5;
  const int lane = threadIdx.x & 31;
  if (row >= kN) return;
  const float* src = (row < kB) ? (zi + (size_t)row * kD)
                                : (zj + (size_t)(row - kB) * kD);
  float4 a = *(const float4*)(src + lane * 8);
  float4 b = *(const float4*)(src + lane * 8 + 4);
  float ss = a.x * a.x + a.y * a.y + a.z * a.z + a.w * a.w +
             b.x * b.x + b.y * b.y + b.z * b.z + b.w * b.w;
#pragma unroll
  for (int m = 1; m < 32; m <<= 1) ss += __shfl_xor(ss, m, 32);
  const float inv = 1.0f / fmaxf(sqrtf(ss), kEps);
  v8h o;
  o[0] = (_Float16)(a.x * inv); o[1] = (_Float16)(a.y * inv);
  o[2] = (_Float16)(a.z * inv); o[3] = (_Float16)(a.w * inv);
  o[4] = (_Float16)(b.x * inv); o[5] = (_Float16)(b.y * inv);
  o[6] = (_Float16)(b.z * inv); o[7] = (_Float16)(b.w * inv);
  *(v8h*)(zh + (size_t)row * kD + lane * 8) = o;
}

// ---------------------------------------------------------------------------
// Kernel 2: per-row O(D) terms, hoisted out of the O(N^2 D) loop:
//   selfexp[i] = exp(2*||zh_i||^2)   (the self-diagonal term to subtract)
//   possim[i]  = 2 * <zh_i, zh_{i^B}> (the positive logit)
// One wave per row.
// ---------------------------------------------------------------------------
__global__ __launch_bounds__(256) void simclr_selfpos(
    const _Float16* __restrict__ zh,
    float* __restrict__ selfexp, float* __restrict__ possim) {
  const int row  = (blockIdx.x * blockDim.x + threadIdx.x) >> 5;
  const int lane = threadIdx.x & 31;
  if (row >= kN) return;
  v8h av = *(const v8h*)(zh + (size_t)row * kD + lane * 8);
  v8h bv = *(const v8h*)(zh + (size_t)(row ^ kB) * kD + lane * 8);
  float ss = 0.0f, pd = 0.0f;
#pragma unroll
  for (int k = 0; k < 8; ++k) {
    const float x = (float)av[k];
    ss += x * x;
    pd += x * (float)bv[k];
  }
#pragma unroll
  for (int m = 1; m < 32; m <<= 1) {
    ss += __shfl_xor(ss, m, 32);
    pd += __shfl_xor(pd, m, 32);
  }
  if (lane == 0) {
    selfexp[row] = __expf(kInvT * ss);
    possim[row]  = kInvT * pd;
  }
}

// ---------------------------------------------------------------------------
// Kernel 3: fused Gram + row-wise sum(exp(sim)) via v_wmma_f32_16x16x32_f16.
// Branchless hot loop, B tiles double-buffered (loads for tile t+1 in flight
// during tile t's MMAs), two independent accumulator chains (even/odd K).
// Row totals INCLUDE the diagonal; it is subtracted in the finalize kernel.
// ---------------------------------------------------------------------------
static __device__ __forceinline__ void load_btile(const _Float16* __restrict__ base,
                                                  v16h* __restrict__ buf) {
#pragma unroll
  for (int c = 0; c < 8; ++c)
    buf[c] = *(const v16h*)(base + 32 * c);   // 16 contiguous halves, 32B aligned
}

static __device__ __forceinline__ void tile_mma(const v16h* __restrict__ afrag,
                                                const v16h* __restrict__ buf,
                                                v8f& esum) {
  v8f acc0 = {}, acc1 = {};
#pragma unroll
  for (int c = 0; c < 8; c += 2) {
    acc0 = __builtin_amdgcn_wmma_f32_16x16x32_f16(false, afrag[c],     false, buf[c],
                                                  (short)0, acc0, false, false);
    acc1 = __builtin_amdgcn_wmma_f32_16x16x32_f16(false, afrag[c + 1], false, buf[c + 1],
                                                  (short)0, acc1, false, false);
  }
#pragma unroll
  for (int v = 0; v < 8; ++v)
    esum[v] += __expf(kInvT * (acc0[v] + acc1[v]));   // |sim|<=2: no max trick
}

__global__ __launch_bounds__(256) void simclr_gram_rowsum(
    const _Float16* __restrict__ zh,
    float* __restrict__ partial) {       // [kChunks][kN]
  const int lane      = threadIdx.x & 31;
  const int wave      = blockIdx.x * (blockDim.x >> 5) + (threadIdx.x >> 5);
  const int rt        = wave / kChunks;       // 0..511 row tile
  const int chunk     = wave % kChunks;       // 0..7
  const int halfSel   = lane >> 4;            // 0 | 1
  const int rowInTile = lane & 15;            // A: row m; B/C/D: col n
  const int aK        = halfSel * 16;         // A K split 0-15 / 16-31
  const int bK        = halfSel * 16;         // B K split 0-15 / 16-31

  // Resident A fragments: K=256 as 8 chunks of 32.
  v16h afrag[8];
  const _Float16* abase = zh + (size_t)(rt * 16 + rowInTile) * kD + aK;
#pragma unroll
  for (int c = 0; c < 8; ++c) {
    afrag[c] = *(const v16h*)(abase + 32 * c);     // K: aK + 0..15
  }

  const _Float16* bbase =
      zh + ((size_t)(chunk * kTilesPerChunk) * 16 + rowInTile) * kD + bK;
  const size_t tileStride = (size_t)16 * kD;       // halves per column tile

  v16h bufA[8], bufB[8];
  v8f  esum = {};

  load_btile(bbase, bufA);                          // tile 0
#pragma unroll 1
  for (int t = 0; t < kTilesPerChunk - 2; t += 2) {
    load_btile(bbase + (size_t)(t + 1) * tileStride, bufB);
    tile_mma(afrag, bufA, esum);                    // tile t
    load_btile(bbase + (size_t)(t + 2) * tileStride, bufA);
    tile_mma(afrag, bufB, esum);                    // tile t+1
  }
  load_btile(bbase + (size_t)(kTilesPerChunk - 1) * tileStride, bufB);
  tile_mma(afrag, bufA, esum);                      // tile 62
  tile_mma(afrag, bufB, esum);                      // tile 63

  // Row sums: reduce across the 16 lanes of each half-wave (xor 1,2,4,8).
#pragma unroll
  for (int v = 0; v < 8; ++v) {
    float s = esum[v];
    s += __shfl_xor(s, 1, 32);
    s += __shfl_xor(s, 2, 32);
    s += __shfl_xor(s, 4, 32);
    s += __shfl_xor(s, 8, 32);
    if (rowInTile == 0)
      partial[chunk * kN + rt * 16 + halfSel * 8 + v] = s;
  }
}

// ---------------------------------------------------------------------------
// Kernel 4: loss = mean_i( log(rowsum_i - exp(sim_ii)) - pos_i ).
// ---------------------------------------------------------------------------
__global__ __launch_bounds__(1024) void simclr_finalize(
    const float* __restrict__ partial, const float* __restrict__ selfexp,
    const float* __restrict__ possim, float* __restrict__ out) {
  __shared__ float sh[1024];
  const int t = threadIdx.x;
  float s = 0.0f;
  for (int r = t; r < kN; r += 1024) {
    float rs = -selfexp[r];
#pragma unroll
    for (int c = 0; c < kChunks; ++c) rs += partial[c * kN + r];
    s += logf(rs) - possim[r];
  }
  sh[t] = s;
  __syncthreads();
  for (int w = 512; w > 0; w >>= 1) {
    if (t < w) sh[t] += sh[t + w];
    __syncthreads();
  }
  if (t == 0) out[0] = sh[0] / (float)kN;
}

// ---------------------------------------------------------------------------
extern "C" void kernel_launch(void* const* d_in, const int* in_sizes, int n_in,
                              void* d_out, int out_size, void* d_ws, size_t ws_size,
                              hipStream_t stream) {
  (void)in_sizes; (void)n_in; (void)out_size; (void)ws_size;
  const float* zi = (const float*)d_in[0];
  const float* zj = (const float*)d_in[1];
  float* out = (float*)d_out;

  // ws: zh f16[N*D] (4MB) | selfexp f32[N] | possim f32[N] | partial f32[8*N]
  char* ws = (char*)d_ws;
  _Float16* zh   = (_Float16*)ws;
  float* selfexp = (float*)(ws + (size_t)kN * kD * sizeof(_Float16));
  float* possim  = selfexp + kN;
  float* partial = possim + kN;

  simclr_normalize_f16<<<dim3((kN * 32) / 256), dim3(256), 0, stream>>>(zi, zj, zh);
  simclr_selfpos<<<dim3((kN * 32) / 256), dim3(256), 0, stream>>>(zh, selfexp, possim);
  // 512 row tiles * 8 chunks = 4096 waves = 512 blocks of 256.
  simclr_gram_rowsum<<<dim3((kTiles * kChunks) / 8), dim3(256), 0, stream>>>(zh, partial);
  simclr_finalize<<<dim3(1), dim3(1024), 0, stream>>>(partial, selfexp, possim, out);
}
